// Model_69647189672235
// MI455X (gfx1250) — matrix operand order repaired and checked by reading, not compiled, not run on hardware
//
#include <hip/hip_runtime.h>
#include <stdint.h>

// Model dims
#define BB   256
#define TT   256
#define DD   72
#define HIDN 512
#define HH   1024
#define H3   3072
#define NPRED 10
#define TOK  (BB * TT)

typedef __attribute__((ext_vector_type(16))) __bf16         v16bf;
typedef __attribute__((ext_vector_type(8)))  float          v8f;
typedef __attribute__((ext_vector_type(16))) unsigned short v16u;
typedef __attribute__((ext_vector_type(8)))  unsigned short v8u;
typedef __attribute__((ext_vector_type(4)))  unsigned int   v4u32;
typedef __attribute__((ext_vector_type(8)))  int            v8i32;
typedef __attribute__((ext_vector_type(4)))  int            v4i32;

union Frag16 { v16u u; v8u h[2]; };

#ifdef __has_builtin
#if __has_builtin(__builtin_amdgcn_tensor_load_to_lds)
#define HAVE_TDM 1
#endif
#endif
#ifndef HAVE_TDM
#define HAVE_TDM 0
#endif

__device__ __forceinline__ unsigned short f2b(float f) {
  unsigned int u = __float_as_uint(f);
  u += 0x7fffu + ((u >> 16) & 1u);            // round-to-nearest-even
  return (unsigned short)(u >> 16);
}
__device__ __forceinline__ float b2f(unsigned short h) {
  return __uint_as_float(((unsigned int)h) << 16);
}
__device__ __forceinline__ float sigmoidf_(float x) { return 1.0f / (1.0f + __expf(-x)); }

// ---------------------------------------------------------------------------
// TDM: async 2-D bf16 tile load (Global -> LDS), tracked by TENSORcnt.
// Descriptor layout per CDNA5 ISA ch.8 (D# groups 0/1; groups 2/3 zero = 2D).
// ---------------------------------------------------------------------------
#if HAVE_TDM
__device__ __forceinline__ void tdm_load_2d_bf16(unsigned lds_off,
                                                 const void* gaddr,
                                                 unsigned tile_d0,  // elems/row
                                                 unsigned tile_d1,  // rows
                                                 unsigned tensor_d0,
                                                 unsigned tensor_d1,
                                                 unsigned stride0) { // elems
  unsigned long long ga = (unsigned long long)(uintptr_t)gaddr;
  v4u32 g0;
  g0[0] = 1u;                                             // count=1, user mode
  g0[1] = lds_off;                                        // lds_addr (bytes)
  g0[2] = (unsigned)(ga & 0xffffffffu);                   // global_addr[31:0]
  g0[3] = (unsigned)((ga >> 32) & 0x01ffffffu) | (2u << 30); // [56:32] | type=2
  v8i32 g1;
  g1[0] = 1 << 16;                                        // data_size=1 (2B)
  g1[1] = (int)((tensor_d0 & 0xffffu) << 16);             // tensor_dim0 lo
  g1[2] = (int)(((tensor_d0 >> 16) & 0xffffu) |
                ((tensor_d1 & 0xffffu) << 16));           // td0 hi | td1 lo
  g1[3] = (int)(((tensor_d1 >> 16) & 0xffffu) |
                ((tile_d0 & 0xffffu) << 16));             // td1 hi | tile_dim0
  g1[4] = (int)(tile_d1 & 0xffffu);                       // tile_dim1 | tile_dim2=0
  g1[5] = (int)stride0;                                   // tensor_dim0_stride lo
  g1[6] = 0;                                              // stride0 hi | stride1 lo
  g1[7] = 0;                                              // stride1 hi
  v4i32 gz = {0, 0, 0, 0};
#if __clang_major__ >= 23
  v8i32 gz8 = {0, 0, 0, 0, 0, 0, 0, 0};
  __builtin_amdgcn_tensor_load_to_lds(g0, g1, gz, gz, gz8, 0);
#else
  __builtin_amdgcn_tensor_load_to_lds(g0, g1, gz, gz, 0);
#endif
}
#endif

// ---------------------------------------------------------------------------
// f32 -> bf16 bulk convert (weights)
// ---------------------------------------------------------------------------
__global__ void k_cvt_bf16(const float* __restrict__ src,
                           unsigned short* __restrict__ dst, int n) {
  int i = blockIdx.x * blockDim.x + threadIdx.x;
  if (i < n) dst[i] = f2b(src[i]);
}

// ---------------------------------------------------------------------------
// Layer-1 of par2lat: Y[row, 0..511] = relu(X[row,0..71] @ W[512,72]^T + b)
// K=72 (not a multiple of 32) -> cheap VALU dot; one block per row.
// ---------------------------------------------------------------------------
__global__ void k_lin1_relu(const float* __restrict__ X,
                            const float* __restrict__ W,   // [512,72]
                            const float* __restrict__ b,   // [512]
                            unsigned short* __restrict__ Y) { // [M,512] bf16
  __shared__ float xs[DD];
  int row = blockIdx.x;
  int tid = threadIdx.x;
  if (tid < DD) xs[tid] = X[(size_t)row * DD + tid];
  __syncthreads();
  for (int n = tid; n < HIDN; n += blockDim.x) {
    float acc = b[n];
    const float* w = W + (size_t)n * DD;
#pragma unroll 8
    for (int k = 0; k < DD; ++k) acc = fmaf(xs[k], w[k], acc);
    Y[(size_t)row * HIDN + n] = f2b(fmaxf(acc, 0.0f));
  }
}

// ---------------------------------------------------------------------------
// Generic bf16 WMMA GEMM: C[M,N] = act(A[M,K] @ W[N,K]^T + bias)
// Block = 256 threads (8 waves, 2x4), tile 64x64; each wave computes a
// 32x16 output (2 accumulators sharing one B fragment). K-step 32, LDS staged.
// M multiple of 64, K multiple of 32; N guarded (Nreal may be 72).
// ---------------------------------------------------------------------------
__global__ void k_gemm_bf16(const unsigned short* __restrict__ A, int K,
                            const unsigned short* __restrict__ W, int Nreal,
                            const float* __restrict__ bias, int relu,
                            unsigned short* __restrict__ Yb,  // [M,Nreal] or null
                            float* __restrict__ Yf) {         // [M,Nreal] or null
  __shared__ unsigned short As[64][32];
  __shared__ unsigned short Ws[64][32];

  const int tid  = threadIdx.x;
  const int lane = tid & 31;
  const int warp = tid >> 5;           // 0..7
  const int wm = warp >> 2;            // 0..1 -> M offset wm*32
  const int wn = warp & 3;             // 0..3 -> N offset wn*16
  const int r = lane & 15;
  const int s = lane >> 4;

  const int mblk = blockIdx.y * 64;
  const int nblk = blockIdx.x * 64;

  v8f acc0 = {}, acc1 = {};

  const int ldrow = tid >> 2;          // 0..63
  const int ldcol = (tid & 3) * 8;     // 0,8,16,24

  for (int k0 = 0; k0 < K; k0 += 32) {
    __syncthreads();
    // stage A tile (64x32 bf16)
    *(v8u*)&As[ldrow][ldcol] =
        *(const v8u*)(A + (size_t)(mblk + ldrow) * K + k0 + ldcol);
    // stage W tile (rows = output columns), zero-pad past Nreal
    int nrow = nblk + ldrow;
    if (nrow < Nreal) {
      *(v8u*)&Ws[ldrow][ldcol] =
          *(const v8u*)(W + (size_t)nrow * K + k0 + ldcol);
    } else {
      v8u z = {0, 0, 0, 0, 0, 0, 0, 0};
      *(v8u*)&Ws[ldrow][ldcol] = z;
    }
    __syncthreads();

    if (k0 + 32 < K)
      __builtin_prefetch(A + (size_t)(mblk + ldrow) * K + k0 + 32 + ldcol, 0, 1);

    Frag16 fa0, fa1, fb;
    fa0.h[0] = *(const v8u*)&As[wm * 32 + r][s * 8];
    fa0.h[1] = *(const v8u*)&As[wm * 32 + r][16 + s * 8];
    fa1.h[0] = *(const v8u*)&As[wm * 32 + 16 + r][s * 8];
    fa1.h[1] = *(const v8u*)&As[wm * 32 + 16 + r][16 + s * 8];
    fb.h[0] = *(const v8u*)&Ws[wn * 16 + r][s * 8];
    fb.h[1] = *(const v8u*)&Ws[wn * 16 + r][16 + s * 8];
    v16bf bmat = __builtin_bit_cast(v16bf, fb.u);

    acc0 = __builtin_amdgcn_wmma_f32_16x16x32_bf16(
        false, __builtin_bit_cast(v16bf, fa0.u), false, bmat, (short)0, acc0, false, false);
    acc1 = __builtin_amdgcn_wmma_f32_16x16x32_bf16(
        false, __builtin_bit_cast(v16bf, fa1.u), false, bmat, (short)0, acc1, false, false);
  }

  const int col = nblk + wn * 16 + r;
  if (col < Nreal) {
    const float bv = bias ? bias[col] : 0.0f;
#pragma unroll
    for (int v = 0; v < 8; ++v) {
      int m0 = mblk + wm * 32 + v + 8 * s;
      int m1 = m0 + 16;
      float y0 = acc0[v] + bv;
      float y1 = acc1[v] + bv;
      if (relu) { y0 = fmaxf(y0, 0.0f); y1 = fmaxf(y1, 0.0f); }
      if (Yb) {
        Yb[(size_t)m0 * Nreal + col] = f2b(y0);
        Yb[(size_t)m1 * Nreal + col] = f2b(y1);
      }
      if (Yf) {
        Yf[(size_t)m0 * Nreal + col] = y0;
        Yf[(size_t)m1 * Nreal + col] = y1;
      }
    }
  }
}

// ---------------------------------------------------------------------------
// Fused GRU step: gh = h @ whh^T (+bhh), then gates with precomputed gi.
// Block = 128 threads / 4 waves; wave w -> 16x16 tile at (blockIdx.y*16,
// blockIdx.x*64 + w*16). Three accumulators (r,z,n) share one A fragment.
// The full 16x1024 bf16 A tile (32KB) is DMA'd into LDS once by the Tensor
// Data Mover (TENSORcnt + workgroup barrier), then the K loop is barrier-free.
// ---------------------------------------------------------------------------
__global__ void k_gru_step(const unsigned short* __restrict__ hprev_bf, // [M,1024]
                           const float* __restrict__ hprev_f,           // [M,1024]
                           const unsigned short* __restrict__ whh,      // [3072,1024] bf16
                           const float* __restrict__ bhh,               // [3072]
                           const unsigned short* __restrict__ gi,       // pre-offset base
                           long gi_stride,
                           float* __restrict__ hnext_f,
                           unsigned short* __restrict__ hnext_bf,
                           unsigned short* __restrict__ seq_out,        // nullable, pre-offset
                           long seq_stride) {
  const int tid  = threadIdx.x;
  const int lane = tid & 31;
  const int warp = tid >> 5;
  const int r = lane & 15;
  const int s = lane >> 4;

  const int m0 = blockIdx.y * 16;
  const int n0 = blockIdx.x * 64 + warp * 16;

  v8f ar = {}, az = {}, an = {};

#if HAVE_TDM
  __shared__ unsigned short As[16 * HH];                 // full A tile, 32KB
  {
    int wid = __builtin_amdgcn_readfirstlane(tid) >> 5;  // uniform -> scalar branch
    if (wid == 0) {
      tdm_load_2d_bf16((unsigned)(uintptr_t)(void*)As,
                       hprev_bf + (size_t)m0 * HH,
                       /*tile_d0=*/HH, /*tile_d1=*/16,
                       /*tensor_d0=*/HH, /*tensor_d1=*/16, /*stride0=*/HH);
      __builtin_amdgcn_s_wait_tensorcnt(0);
    }
  }
  __syncthreads();

  for (int k0 = 0; k0 < HH; k0 += 32) {
    Frag16 fa;
    fa.h[0] = *(const v8u*)&As[r * HH + k0 + s * 8];
    fa.h[1] = *(const v8u*)&As[r * HH + k0 + 16 + s * 8];
    v16bf a = __builtin_bit_cast(v16bf, fa.u);
#else
  __shared__ unsigned short As[16][32];
  for (int k0 = 0; k0 < HH; k0 += 32) {
    __syncthreads();
    if (tid < 64) {
      int row = tid >> 2, c = (tid & 3) * 8;
      *(v8u*)&As[row][c] =
          *(const v8u*)(hprev_bf + (size_t)(m0 + row) * HH + k0 + c);
    }
    __syncthreads();
    Frag16 fa;
    fa.h[0] = *(const v8u*)&As[r][s * 8];
    fa.h[1] = *(const v8u*)&As[r][16 + s * 8];
    v16bf a = __builtin_bit_cast(v16bf, fa.u);
#endif

    Frag16 fr, fz, fn;
    const unsigned short* wr = whh + (size_t)(0 * HH + n0 + r) * HH + k0 + s * 8;
    const unsigned short* wz = whh + (size_t)(1 * HH + n0 + r) * HH + k0 + s * 8;
    const unsigned short* wn = whh + (size_t)(2 * HH + n0 + r) * HH + k0 + s * 8;
    fr.h[0] = *(const v8u*)(wr);  fr.h[1] = *(const v8u*)(wr + 16);
    fz.h[0] = *(const v8u*)(wz);  fz.h[1] = *(const v8u*)(wz + 16);
    fn.h[0] = *(const v8u*)(wn);  fn.h[1] = *(const v8u*)(wn + 16);

    ar = __builtin_amdgcn_wmma_f32_16x16x32_bf16(
        false, a, false, __builtin_bit_cast(v16bf, fr.u), (short)0, ar, false, false);
    az = __builtin_amdgcn_wmma_f32_16x16x32_bf16(
        false, a, false, __builtin_bit_cast(v16bf, fz.u), (short)0, az, false, false);
    an = __builtin_amdgcn_wmma_f32_16x16x32_bf16(
        false, a, false, __builtin_bit_cast(v16bf, fn.u), (short)0, an, false, false);
  }

  const int col = n0 + r;
  const float br = bhh[col], bz = bhh[HH + col], bn = bhh[2 * HH + col];
#pragma unroll
  for (int v = 0; v < 8; ++v) {
    int m = m0 + v + 8 * s;
    const unsigned short* g = gi + (size_t)m * gi_stride;
    float gir = b2f(g[col]);
    float giz = b2f(g[HH + col]);
    float gin = b2f(g[2 * HH + col]);
    float rr = sigmoidf_(gir + ar[v] + br);
    float zz = sigmoidf_(giz + az[v] + bz);
    float nn = tanhf(gin + rr * (an[v] + bn));
    float hp = hprev_f[(size_t)m * HH + col];
    float hn = (1.0f - zz) * nn + zz * hp;
    hnext_f[(size_t)m * HH + col] = hn;
    unsigned short hb = f2b(hn);
    hnext_bf[(size_t)m * HH + col] = hb;
    if (seq_out) seq_out[(size_t)m * seq_stride + col] = hb;
  }
}

// ---------------------------------------------------------------------------
// Attention logits: logits[tok] = att_h[tok,:512] . w2 + b2  (one wave/token)
// ---------------------------------------------------------------------------
__global__ void k_att_logits(const unsigned short* __restrict__ ah,  // [TOK,512] bf16
                             const float* __restrict__ w2,            // [512]
                             const float* __restrict__ b2,            // [1]
                             float* __restrict__ logits, int ntok) {
  int wid  = (blockIdx.x * blockDim.x + threadIdx.x) >> 5;
  int lane = threadIdx.x & 31;
  if (wid >= ntok) return;
  const unsigned short* row = ah + (size_t)wid * HIDN;
  float acc = 0.0f;
#pragma unroll 4
  for (int k = lane; k < HIDN; k += 32) acc += b2f(row[k]) * w2[k];
#pragma unroll
  for (int off = 16; off > 0; off >>= 1) acc += __shfl_xor(acc, off, 32);
  if (lane == 0) logits[wid] = acc + b2[0];
}

// ---------------------------------------------------------------------------
// Softmax over T + weighted pooling -> Rep. One block (256 thr) per batch b.
// ---------------------------------------------------------------------------
__global__ void k_softmax_rep(const float* __restrict__ logits,        // [B*T]
                              const unsigned short* __restrict__ outbf,// [B*T,1024]
                              float* __restrict__ rep_out,             // [B,1024] (d_out)
                              float* __restrict__ hdec_f,              // [B,1024]
                              unsigned short* __restrict__ hdec_bf) {  // [B,1024]
  __shared__ float red[TT];
  __shared__ float wts[TT];
  const int b = blockIdx.x, tid = threadIdx.x;
  float v = logits[(size_t)b * TT + tid];
  red[tid] = v;
  __syncthreads();
  for (int s2 = 128; s2 > 0; s2 >>= 1) {
    if (tid < s2) red[tid] = fmaxf(red[tid], red[tid + s2]);
    __syncthreads();
  }
  float mx = red[0];
  __syncthreads();
  float e = __expf(v - mx);
  red[tid] = e;
  wts[tid] = e;
  __syncthreads();
  for (int s2 = 128; s2 > 0; s2 >>= 1) {
    if (tid < s2) red[tid] += red[tid + s2];
    __syncthreads();
  }
  float inv = 1.0f / red[0];

  float a0 = 0, a1 = 0, a2 = 0, a3 = 0;
  const int h = tid * 4;
  const unsigned short* base = outbf + (size_t)b * TT * HH + h;
  for (int t = 0; t < TT; ++t) {
    float wt = wts[t] * inv;
    const unsigned short* rowp = base + (size_t)t * HH;
    a0 += wt * b2f(rowp[0]);
    a1 += wt * b2f(rowp[1]);
    a2 += wt * b2f(rowp[2]);
    a3 += wt * b2f(rowp[3]);
  }
  size_t o = (size_t)b * HH + h;
  rep_out[o + 0] = a0; rep_out[o + 1] = a1; rep_out[o + 2] = a2; rep_out[o + 3] = a3;
  hdec_f[o + 0] = a0;  hdec_f[o + 1] = a1;  hdec_f[o + 2] = a2;  hdec_f[o + 3] = a3;
  hdec_bf[o + 0] = f2b(a0); hdec_bf[o + 1] = f2b(a1);
  hdec_bf[o + 2] = f2b(a2); hdec_bf[o + 3] = f2b(a3);
}

// ---------------------------------------------------------------------------
__global__ void k_gather_last(const float* __restrict__ x, float* __restrict__ p) {
  int i = blockIdx.x * blockDim.x + threadIdx.x;
  if (i >= BB * DD) return;
  int b = i / DD, d = i - b * DD;
  p[i] = x[((size_t)b * TT + (TT - 1)) * DD + d];
}

// ---------------------------------------------------------------------------
__global__ void k_scatter_pred(const float* __restrict__ p, float* __restrict__ out,
                               int step) {
  int i = blockIdx.x * blockDim.x + threadIdx.x;
  if (i >= BB * DD) return;
  int b = i / DD, d = i - b * DD;
  out[((size_t)b * NPRED + step) * DD + d] = p[i];
}

// ===========================================================================
extern "C" void kernel_launch(void* const* d_in, const int* in_sizes, int n_in,
                              void* d_out, int out_size, void* d_ws, size_t ws_size,
                              hipStream_t stream) {
  (void)in_sizes; (void)n_in; (void)out_size; (void)ws_size;

  const float* x        = (const float*)d_in[0];
  // d_in[1] = to_predict (device scalar) -> hardcoded NPRED
  const float* p2l_w1   = (const float*)d_in[2];
  const float* p2l_b1   = (const float*)d_in[3];
  const float* p2l_w2   = (const float*)d_in[4];
  const float* p2l_b2   = (const float*)d_in[5];
  const float* enc_wih  = (const float*)d_in[6];
  const float* enc_whh  = (const float*)d_in[7];
  const float* enc_bih  = (const float*)d_in[8];
  const float* enc_bhh  = (const float*)d_in[9];
  const float* dec_wih  = (const float*)d_in[10];
  const float* dec_whh  = (const float*)d_in[11];
  const float* dec_bih  = (const float*)d_in[12];
  const float* dec_bhh  = (const float*)d_in[13];
  const float* l2p_w1   = (const float*)d_in[14];
  const float* l2p_b1   = (const float*)d_in[15];
  const float* l2p_w2   = (const float*)d_in[16];
  const float* l2p_b2   = (const float*)d_in[17];
  const float* l2s_w1   = (const float*)d_in[18];
  const float* l2s_b1   = (const float*)d_in[19];
  const float* l2s_w2   = (const float*)d_in[20];
  const float* l2s_b2   = (const float*)d_in[21];

  // --- workspace carve (256B aligned bump allocator) ---
  char* base = (char*)d_ws;
  size_t off = 0;
  auto alloc = [&](size_t bytes) -> void* {
    void* p = base + off;
    off = (off + bytes + 255) & ~(size_t)255;
    return p;
  };
  unsigned short* wb_p2l_w2  = (unsigned short*)alloc((size_t)HH * HIDN * 2);
  unsigned short* wb_enc_wih = (unsigned short*)alloc((size_t)H3 * HH * 2);
  unsigned short* wb_enc_whh = (unsigned short*)alloc((size_t)H3 * HH * 2);
  unsigned short* wb_dec_wih = (unsigned short*)alloc((size_t)H3 * HH * 2);
  unsigned short* wb_dec_whh = (unsigned short*)alloc((size_t)H3 * HH * 2);
  unsigned short* wb_l2p_w1  = (unsigned short*)alloc((size_t)HIDN * HH * 2);
  unsigned short* wb_l2p_w2  = (unsigned short*)alloc((size_t)DD * HIDN * 2);
  unsigned short* wb_l2s_w1  = (unsigned short*)alloc((size_t)HIDN * HH * 2);

  unsigned short* h1_bf     = (unsigned short*)alloc((size_t)TOK * HIDN * 2); // layer1 out / att_h
  unsigned short* latent_bf = (unsigned short*)alloc((size_t)TOK * HH * 2);
  unsigned short* xp_bf     = (unsigned short*)alloc((size_t)TOK * H3 * 2);
  unsigned short* out_bf    = (unsigned short*)alloc((size_t)TOK * HH * 2);
  float*          logits    = (float*)alloc((size_t)TOK * 4);

  float*          hA_f  = (float*)alloc((size_t)BB * HH * 4);
  float*          hB_f  = (float*)alloc((size_t)BB * HH * 4);
  unsigned short* hA_bf = (unsigned short*)alloc((size_t)BB * HH * 2);
  unsigned short* hB_bf = (unsigned short*)alloc((size_t)BB * HH * 2);

  unsigned short* gi_dec  = (unsigned short*)alloc((size_t)BB * H3 * 2);
  unsigned short* l1_dec  = (unsigned short*)alloc((size_t)BB * HIDN * 2);
  unsigned short* inp_dec = (unsigned short*)alloc((size_t)BB * HH * 2);
  unsigned short* hid_dec = (unsigned short*)alloc((size_t)BB * HIDN * 2);
  float*          pA      = (float*)alloc((size_t)BB * DD * 4);
  float*          pB      = (float*)alloc((size_t)BB * DD * 4);

  float* out_preds = (float*)d_out;                           // [B, NPRED, 72]
  float* out_rep   = (float*)d_out + (size_t)BB * NPRED * DD; // [B, 1024]

  auto cvt = [&](const float* s, unsigned short* d, int n) {
    k_cvt_bf16<<<(n + 255) / 256, 256, 0, stream>>>(s, d, n);
  };

  // --- weight conversions (f32 -> bf16) ---
  cvt(p2l_w2,  wb_p2l_w2,  HH * HIDN);
  cvt(enc_wih, wb_enc_wih, H3 * HH);
  cvt(enc_whh, wb_enc_whh, H3 * HH);
  cvt(dec_wih, wb_dec_wih, H3 * HH);
  cvt(dec_whh, wb_dec_whh, H3 * HH);
  cvt(l2p_w1,  wb_l2p_w1,  HIDN * HH);
  cvt(l2p_w2,  wb_l2p_w2,  DD * HIDN);
  cvt(l2s_w1,  wb_l2s_w1,  HIDN * HH);

  // --- par2lat over all tokens ---
  k_lin1_relu<<<TOK, 256, 0, stream>>>(x, p2l_w1, p2l_b1, h1_bf);
  k_gemm_bf16<<<dim3(HH / 64, TOK / 64), 256, 0, stream>>>(
      h1_bf, HIDN, wb_p2l_w2, HH, p2l_b2, 1, latent_bf, nullptr);

  // --- xp = latent @ enc_wih^T + bih  (input projections for the scan) ---
  k_gemm_bf16<<<dim3(H3 / 64, TOK / 64), 256, 0, stream>>>(
      latent_bf, HH, wb_enc_wih, H3, enc_bih, 0, xp_bf, nullptr);

  // --- encoder GRU scan ---
  hipMemsetAsync(hA_f, 0, (size_t)BB * HH * 4, stream);
  hipMemsetAsync(hA_bf, 0, (size_t)BB * HH * 2, stream);
  float* hc_f = hA_f;  float* hn_f = hB_f;
  unsigned short* hc_b = hA_bf; unsigned short* hn_b = hB_bf;
  for (int t = 0; t < TT; ++t) {
    k_gru_step<<<dim3(HH / 64, BB / 16), 128, 0, stream>>>(
        hc_b, hc_f, wb_enc_whh, enc_bhh,
        xp_bf + (size_t)t * H3, (long)TT * H3,
        hn_f, hn_b,
        out_bf + (size_t)t * HH, (long)TT * HH);
    { float* tf = hc_f; hc_f = hn_f; hn_f = tf; }
    { unsigned short* tb = hc_b; hc_b = hn_b; hn_b = tb; }
  }

  // --- attention pooling ---
  k_gemm_bf16<<<dim3(HIDN / 64, TOK / 64), 256, 0, stream>>>(
      out_bf, HH, wb_l2s_w1, HIDN, l2s_b1, 1, h1_bf, nullptr); // att hidden (reuse h1)
  k_att_logits<<<TOK / 8, 256, 0, stream>>>(h1_bf, l2s_w2, l2s_b2, logits, TOK);
  k_softmax_rep<<<BB, TT, 0, stream>>>(logits, out_bf, out_rep, hA_f, hA_bf);

  // --- autoregressive decoder ---
  k_gather_last<<<(BB * DD + 255) / 256, 256, 0, stream>>>(x, pA);
  float* p_cur = pA;  float* p_nxt = pB;
  hc_f = hA_f; hn_f = hB_f; hc_b = hA_bf; hn_b = hB_bf;
  for (int step = 0; step < NPRED; ++step) {
    k_lin1_relu<<<BB, 256, 0, stream>>>(p_cur, p2l_w1, p2l_b1, l1_dec);
    k_gemm_bf16<<<dim3(HH / 64, BB / 64), 256, 0, stream>>>(
        l1_dec, HIDN, wb_p2l_w2, HH, p2l_b2, 1, inp_dec, nullptr);
    k_gemm_bf16<<<dim3(H3 / 64, BB / 64), 256, 0, stream>>>(
        inp_dec, HH, wb_dec_wih, H3, dec_bih, 0, gi_dec, nullptr);
    k_gru_step<<<dim3(HH / 64, BB / 16), 128, 0, stream>>>(
        hc_b, hc_f, wb_dec_whh, dec_bhh, gi_dec, (long)H3,
        hn_f, hn_b, nullptr, 0);
    k_gemm_bf16<<<dim3(HIDN / 64, BB / 64), 256, 0, stream>>>(
        hn_b, HH, wb_l2p_w1, HIDN, l2p_b1, 1, hid_dec, nullptr);
    k_gemm_bf16<<<dim3((DD + 63) / 64, BB / 64), 256, 0, stream>>>(
        hid_dec, HIDN, wb_l2p_w2, DD, l2p_b2, 0, nullptr, p_nxt);
    k_scatter_pred<<<(BB * DD + 255) / 256, 256, 0, stream>>>(p_nxt, out_preds, step);
    { float* tf = p_cur; p_cur = p_nxt; p_nxt = tf; }
    { float* tf = hc_f; hc_f = hn_f; hn_f = tf; }
    { unsigned short* tb = hc_b; hc_b = hn_b; hn_b = tb; }
  }
}
